// GCNLayer_63788854281032
// MI455X (gfx1250) — compile-verified, matrix operand order checked by reading, not run on hardware
//
#include <hip/hip_runtime.h>

typedef __bf16 bf16;
typedef __bf16 v8bf  __attribute__((ext_vector_type(8)));
typedef __bf16 v16bf __attribute__((ext_vector_type(16)));
typedef float  v8f   __attribute__((ext_vector_type(8)));

#define TILE_K  64
#define LDS_STR 72   // 64 K + 8 pad (bf16 elements) per row

// low 32 bits of a generic pointer to __shared__ = wave-relative LDS byte address
__device__ __forceinline__ unsigned lds_addr32(const void* p) {
  return (unsigned)(size_t)p;
}

// CDNA5 async copy: memory -> LDS, no VGPR round trip, tracked by ASYNCcnt.
// INST_OFFSET applies to both the LDS and global addresses (ISA 15.18.3).
#define ASYNC_COPY_B128(ldsoff, gptr, IMM)                                     \
  asm volatile("global_load_async_to_lds_b128 %0, %1, off offset:" #IMM        \
               :: "v"(ldsoff), "v"(gptr) : "memory")

#define WAIT_ASYNCCNT0() asm volatile("s_wait_asynccnt 0x0" ::: "memory")

// -------- transpose + fp32->bf16 convert: src[R][C] f32 -> dst[C][R] bf16 ----
__global__ __launch_bounds__(256) void xpose_cvt_kernel(
    const float* __restrict__ src, bf16* __restrict__ dst, int R, int C)
{
  __shared__ bf16 tile[64][64 + 8];
  const int t  = threadIdx.x;
  const int gr = blockIdx.y * 64;
  const int gc = blockIdx.x * 64;

  {
    const int lr  = t >> 2;
    const int lc0 = (t & 3) * 16;
    const float4* p =
        reinterpret_cast<const float4*>(src + (size_t)(gr + lr) * C + gc + lc0);
    #pragma unroll
    for (int i = 0; i < 4; ++i) {
      float4 v = p[i];
      tile[lr][lc0 + 4*i + 0] = (bf16)v.x;
      tile[lr][lc0 + 4*i + 1] = (bf16)v.y;
      tile[lr][lc0 + 4*i + 2] = (bf16)v.z;
      tile[lr][lc0 + 4*i + 3] = (bf16)v.w;
    }
  }
  __syncthreads();
  {
    const int lc  = t >> 2;
    const int lr0 = (t & 3) * 16;
    alignas(32) bf16 vals[16];
    #pragma unroll
    for (int i = 0; i < 16; ++i) vals[i] = tile[lr0 + i][lc];
    uint4* q = reinterpret_cast<uint4*>(dst + (size_t)(gc + lc) * R + gr + lr0);
    const uint4* s = reinterpret_cast<const uint4*>(vals);
    q[0] = s[0];
    q[1] = s[1];
  }
}

// -------- C[M x 256] = A[M x K](f32) @ B[K x 256]   (B given transposed bf16) ----
// 256 threads = 8 wave32 (2M x 4N); block tile 32 x 256; K-tile 64 (2 WMMA K-steps).
// A: global f32 -> VGPR cvt -> LDS (bandwidth-critical adj stream, read once as f32).
// B: global bf16 -> LDS via GLOBAL_LOAD_ASYNC_TO_LDS_B128 (ASYNCcnt path).
__global__ __launch_bounds__(256) void gemm_bf16_wmma_kernel(
    const float* __restrict__ A, long lda, int K,
    const bf16* __restrict__ BT,           // [256][K] bf16 (col-major of B)
    float* __restrict__ C, int do_relu)
{
  __shared__ bf16 lA[32][LDS_STR];   // 32 rows  x 64 K
  __shared__ bf16 lB[256][LDS_STR];  // 256 cols x 64 K

  const int t    = threadIdx.x;
  const int lane = t & 31;
  const int w    = t >> 5;
  const int wm   = w >> 2;      // 0..1  (M strip)
  const int wn   = w & 3;       // 0..3  (N quarter)
  const int hk   = lane >> 4;   // K-half selector per ISA 16-bit layout
  const int lm   = lane & 15;
  const long row0 = (long)blockIdx.x * 32;

  v8f acc[4];
  #pragma unroll
  for (int i = 0; i < 4; ++i)
    #pragma unroll
    for (int e = 0; e < 8; ++e) acc[i][e] = 0.0f;

  const int ar = t >> 3;        // 0..31 : A row staged by this thread
  const int aj = (t & 7) * 8;   // 0..56 : 8-float K segment
  const float* aptr = A + (row0 + ar) * lda + aj;
  const bf16*  bptr = BT + (size_t)t * K;           // thread t owns column t
  const unsigned ldsB = lds_addr32(&lB[t][0]);      // 128B contiguous dest
  const int ra = wm * 16 + lm;

  for (int k0 = 0; k0 < K; k0 += TILE_K) {
    if (k0 + 2 * TILE_K <= K)                       // keep the adj stream ahead
      __builtin_prefetch(aptr + k0 + 2 * TILE_K, 0, 1);

    { // B tile: 128 bytes/thread, async straight into LDS
      const bf16* g = bptr + k0;
      ASYNC_COPY_B128(ldsB, g, 0);
      ASYNC_COPY_B128(ldsB, g, 16);
      ASYNC_COPY_B128(ldsB, g, 32);
      ASYNC_COPY_B128(ldsB, g, 48);
      ASYNC_COPY_B128(ldsB, g, 64);
      ASYNC_COPY_B128(ldsB, g, 80);
      ASYNC_COPY_B128(ldsB, g, 96);
      ASYNC_COPY_B128(ldsB, g, 112);
    }
    { // A tile: fp32 -> bf16 in-register, one b128 LDS store
      float4 v0 = *reinterpret_cast<const float4*>(aptr + k0);
      float4 v1 = *reinterpret_cast<const float4*>(aptr + k0 + 4);
      alignas(16) bf16 ab[8];
      ab[0] = (bf16)v0.x; ab[1] = (bf16)v0.y; ab[2] = (bf16)v0.z; ab[3] = (bf16)v0.w;
      ab[4] = (bf16)v1.x; ab[5] = (bf16)v1.y; ab[6] = (bf16)v1.z; ab[7] = (bf16)v1.w;
      *reinterpret_cast<uint4*>(&lA[ar][aj]) = *reinterpret_cast<const uint4*>(ab);
    }
    WAIT_ASYNCCNT0();          // LDS writes from async path visible
    __syncthreads();           // + dscnt for the A store

    #pragma unroll
    for (int s = 0; s < 2; ++s) {          // two 16x16x32 K-steps per tile
      const int kb = s * 32 + hk * 8;
      v16bf afrag;
      {
        v8bf alo = *reinterpret_cast<const v8bf*>(&lA[ra][kb]);
        v8bf ahi = *reinterpret_cast<const v8bf*>(&lA[ra][kb + 16]);
        #pragma unroll
        for (int e = 0; e < 8; ++e) { afrag[e] = alo[e]; afrag[e + 8] = ahi[e]; }
      }
      v16bf bfrag[4];
      #pragma unroll
      for (int nt = 0; nt < 4; ++nt) {     // batch fragment loads first
        const int cb = wn * 64 + nt * 16 + lm;
        v8bf blo = *reinterpret_cast<const v8bf*>(&lB[cb][kb]);
        v8bf bhi = *reinterpret_cast<const v8bf*>(&lB[cb][kb + 16]);
        #pragma unroll
        for (int e = 0; e < 8; ++e) { bfrag[nt][e] = blo[e]; bfrag[nt][e + 8] = bhi[e]; }
      }
      #pragma unroll
      for (int nt = 0; nt < 4; ++nt)       // then a clean WMMA burst
        acc[nt] = __builtin_amdgcn_wmma_f32_16x16x32_bf16(
            false, afrag, false, bfrag[nt], (short)0, acc[nt], false, false);
    }
    __syncthreads();
  }

  // Epilogue: C/D layout -> row = i + hk*8 (per VGPR), col = lane&15
  #pragma unroll
  for (int nt = 0; nt < 4; ++nt) {
    const int col = wn * 64 + nt * 16 + lm;
    #pragma unroll
    for (int i = 0; i < 8; ++i) {
      const long row = row0 + wm * 16 + hk * 8 + i;
      float v = acc[nt][i];
      if (do_relu) v = fmaxf(v, 0.0f);
      C[row * 256 + col] = v;
    }
  }
}

extern "C" void kernel_launch(void* const* d_in, const int* in_sizes, int n_in,
                              void* d_out, int out_size, void* d_ws, size_t ws_size,
                              hipStream_t stream)
{
  const float* data = (const float*)d_in[0];   // [8192][256]  f32
  const float* adj  = (const float*)d_in[1];   // [8192][8192] f32
  const float* W    = (const float*)d_in[2];   // [256][256]   f32
  float* out = (float*)d_out;                  // [8192][256]  f32

  char* ws = (char*)d_ws;
  bf16*  dataT = (bf16*)(ws);                  // [256][8192] bf16  (4 MB)
  bf16*  WT    = (bf16*)(ws + (4ull << 20));   // [256][256]  bf16  (128 KB)
  float* af    = (float*)(ws + (5ull << 20));  // [8192][256] f32   (8 MB)

  dim3 tb(256);
  // dataT[c][k] = data[k][c] ; WT[c][k] = W[k][c]   (bf16)
  xpose_cvt_kernel<<<dim3(256 / 64, 8192 / 64), tb, 0, stream>>>(data, dataT, 8192, 256);
  xpose_cvt_kernel<<<dim3(4, 4),                tb, 0, stream>>>(W,    WT,    256,  256);
  // af = adj @ data          (HBM-bound on 256 MB adj stream)
  gemm_bf16_wmma_kernel<<<dim3(8192 / 32), tb, 0, stream>>>(adj, 8192, 8192, dataT, af, 0);
  // out = relu(af @ W)
  gemm_bf16_wmma_kernel<<<dim3(8192 / 32), tb, 0, stream>>>(af, 256, 256, WT, out, 1);
}
